// EfficientAttention_88562225643974
// MI455X (gfx1250) — matrix operand order repaired and checked by reading, not compiled, not run on hardware
//
#include <hip/hip_runtime.h>
#include <hip/hip_bf16.h>

typedef __attribute__((ext_vector_type(16))) _Float16 v16h;
typedef __attribute__((ext_vector_type(8)))  _Float16 v8h;
typedef __attribute__((ext_vector_type(4)))  _Float16 v4h;
typedef __attribute__((ext_vector_type(8)))  float    v8f;

#define EA_B 8
#define EA_S 4096
#define EA_H 16
#define EA_D 128

// ---------------------------------------------------------------------------
// Kernel 1: per-(b,h,d) online softmax stats over S for K.
// grid = B*H blocks, 1024 threads (8 s-groups x 128 d). Coalesced 512B rows.
// ---------------------------------------------------------------------------
__global__ void __launch_bounds__(1024)
ea_stats(const float* __restrict__ kg,
         float* __restrict__ smax, float* __restrict__ sinv) {
    const int bh = blockIdx.x;
    const int b = bh >> 4, h = bh & 15;
    const int t = threadIdx.x;
    const int d = t & 127;
    const int g = t >> 7;                       // 0..7
    const float* base = kg + ((size_t)b * EA_S * EA_H + h) * EA_D + d;

    float m = -3.0e38f, l = 0.0f;
    for (int s = g; s < EA_S; s += 8) {
        float x = base[(size_t)s * (EA_H * EA_D)];
        float mn = fmaxf(m, x);
        l = l * __expf(m - mn) + __expf(x - mn);
        m = mn;
    }
    __shared__ float mp[1024];
    __shared__ float lp[1024];
    mp[t] = m; lp[t] = l;
    __syncthreads();
    if (t < 128) {
        float M = -3.0e38f;
        #pragma unroll
        for (int i = 0; i < 8; ++i) M = fmaxf(M, mp[i * 128 + t]);
        float L = 0.0f;
        #pragma unroll
        for (int i = 0; i < 8; ++i) L += lp[i * 128 + t] * __expf(mp[i * 128 + t] - M);
        smax[bh * 128 + t] = M;
        sinv[bh * 128 + t] = 1.0f / L;
    }
}

// ---------------------------------------------------------------------------
// Kernel 2: context[bh][d][e] = sum_s softmaxK[s][d] * v[s][e]   (128x128)
// grid = B*H blocks, 256 threads = 8 waves; wave w owns rows d = 16w..16w+15.
// S chunked by 32; each thread owns one d column and a contiguous 16-wide
// s-range so the transposed f16 LDS fill is 4x ds_store_b128 per thread.
// All 8 B fragments are loaded before the WMMA chain so the 8 WMMAs issue
// back-to-back instead of load->wait->wmma serialization.
// ---------------------------------------------------------------------------
__global__ void __launch_bounds__(256)
ea_context(const float* __restrict__ kg, const float* __restrict__ vg,
           const float* __restrict__ smax, const float* __restrict__ sinv,
           float* __restrict__ ctx) {
    const int bh = blockIdx.x;
    const int b = bh >> 4, h = bh & 15;
    const int t = threadIdx.x;
    const int lane = t & 31, wave = t >> 5;
    const int laneM = lane & 15;
    const int hiSel = lane >> 4;                // 0: lanes 0-15, 1: lanes 16-31
    const int m0 = wave * 16;

    __shared__ __align__(16) _Float16 kb[128 * 32];   // A tile: [d][s_local]
    __shared__ __align__(16) _Float16 vb[128 * 32];   // B tile: [e][s_local]

    const int dl   = t & 127;                   // this thread's fixed column
    const int soff = (t >> 7) * 16;             // contiguous 16-wide s range
    const float kmax = smax[bh * 128 + dl];
    const float kinv = sinv[bh * 128 + dl];
    const size_t rs = EA_H * EA_D;              // 2048 floats between s rows
    const float* kbase = kg + ((size_t)b * EA_S * EA_H + h) * EA_D + dl;
    const float* vbase = vg + ((size_t)b * EA_S * EA_H + h) * EA_D + dl;

    v8f acc[8] = {};

    for (int s0 = 0; s0 < EA_S; s0 += 32) {
        // Global: per i, each 128-thread group reads one full 512B row.
        v8h kp0, kp1, vp0, vp1;
        #pragma unroll
        for (int i = 0; i < 16; ++i) {
            int sl = soff + i;
            float kx = kbase[(size_t)(s0 + sl) * rs];
            float vx = vbase[(size_t)(s0 + sl) * rs];
            _Float16 kf = (_Float16)(__expf(kx - kmax) * kinv);
            _Float16 vf = (_Float16)vx;
            if (i < 8) { kp0[i] = kf; vp0[i] = vf; }
            else       { kp1[i - 8] = kf; vp1[i - 8] = vf; }
        }
        *(v8h*)&kb[dl * 32 + soff]     = kp0;
        *(v8h*)&kb[dl * 32 + soff + 8] = kp1;
        *(v8h*)&vb[dl * 32 + soff]     = vp0;
        *(v8h*)&vb[dl * 32 + soff + 8] = vp1;
        __syncthreads();

        // A fragment: lane row M = m0+laneM; halves K = {0..15} (lo) /
        // {16..31} (hi lanes) -> two contiguous 16B LDS loads.
        const _Float16* arow = &kb[(m0 + laneM) * 32];
        v8h alo = *(const v8h*)(arow + hiSel * 16);
        v8h ahi = *(const v8h*)(arow + hiSel * 16 + 8);
        v16h afrag;
        #pragma unroll
        for (int i = 0; i < 8; ++i) { afrag[i] = alo[i]; afrag[i + 8] = ahi[i]; }

        // Load ALL 8 B fragments first (one ds_load clause), then chain WMMAs.
        v16h bfr[8];
        #pragma unroll
        for (int nt = 0; nt < 8; ++nt) {
            const _Float16* brow = &vb[(nt * 16 + laneM) * 32 + hiSel * 16];
            v8h b0 = *(const v8h*)(brow);
            v8h b1 = *(const v8h*)(brow + 8);
            #pragma unroll
            for (int i = 0; i < 8; ++i) { bfr[nt][i] = b0[i]; bfr[nt][i + 8] = b1[i]; }
        }
        #pragma unroll
        for (int nt = 0; nt < 8; ++nt) {
            acc[nt] = __builtin_amdgcn_wmma_f32_16x16x32_f16(
                false, afrag, false, bfr[nt], (short)0, acc[nt], false, false);
        }
        __syncthreads();
    }

    // C layout: VGPR r -> (M = r (+8 for hi lanes), N = laneM)
    float* cbase = ctx + (size_t)bh * 128 * 128;
    #pragma unroll
    for (int nt = 0; nt < 8; ++nt) {
        #pragma unroll
        for (int r = 0; r < 8; ++r) {
            int row = m0 + r + hiSel * 8;
            cbase[row * 128 + nt * 16 + laneM] = acc[nt][r];
        }
    }
}

// ---------------------------------------------------------------------------
// Kernel 3: out[s][e] = sum_d softmaxQ[s][d] * context[d][e]
// grid = (32 s-blocks, B*H), 256 threads = 8 waves; fused Q row softmax via
// wave-shuffle reductions, then 4 K-chunks x 8 N-tiles of WMMA per wave.
// ---------------------------------------------------------------------------
__global__ void __launch_bounds__(256)
ea_out(const float* __restrict__ qg, const float* __restrict__ ctx,
       float* __restrict__ outg) {
    const int bh = blockIdx.y;
    const int b = bh >> 4, h = bh & 15;
    const int s0 = blockIdx.x * 128;
    const int t = threadIdx.x;
    const int lane = t & 31, wave = t >> 5;
    const int laneM = lane & 15;
    const int hiSel = lane >> 4;
    const int m0 = wave * 16;

    __shared__ __align__(16) _Float16 qb[128 * 128];  // A: [s_local][d]  32KB
    __shared__ __align__(16) _Float16 cb[128 * 128];  // B: [e][d]        32KB

    // context (L2-resident, 8MB total) -> transposed f16.
    // thread = (e, d-group): global reads stay row-coalesced (consecutive e
    // at fixed d), LDS writes are contiguous 64-element d runs per thread.
    {
        const float* cg = ctx + (size_t)bh * 128 * 128;
        const int e_ = t & 127;
        const int dg = (t >> 7) * 64;           // 0 or 64
        #pragma unroll
        for (int c = 0; c < 8; ++c) {
            v8h cv;
            #pragma unroll
            for (int j = 0; j < 8; ++j) {
                int d_ = dg + c * 8 + j;
                cv[j] = (_Float16)cg[d_ * 128 + e_];
            }
            *(v8h*)&cb[e_ * 128 + dg + c * 8] = cv;
        }
    }

    // Q rows: one wave per row, coalesced float4 loads, shuffle reductions,
    // packed v4h (ds_store_b64) writeback.
    const float* qbase = qg + (((size_t)b * EA_S + s0) * EA_H + h) * EA_D;
    for (int r = wave; r < 128; r += 8) {
        const float4* qrow = (const float4*)(qbase + (size_t)r * (EA_H * EA_D));
        float4 xv = qrow[lane];
        float m = fmaxf(fmaxf(xv.x, xv.y), fmaxf(xv.z, xv.w));
        #pragma unroll
        for (int off = 16; off > 0; off >>= 1) m = fmaxf(m, __shfl_xor(m, off));
        float x0 = __expf(xv.x - m), x1 = __expf(xv.y - m);
        float x2 = __expf(xv.z - m), x3 = __expf(xv.w - m);
        float sum = x0 + x1 + x2 + x3;
        #pragma unroll
        for (int off = 16; off > 0; off >>= 1) sum += __shfl_xor(sum, off);
        float inv = 1.0f / sum;
        v4h qv;
        qv[0] = (_Float16)(x0 * inv); qv[1] = (_Float16)(x1 * inv);
        qv[2] = (_Float16)(x2 * inv); qv[3] = (_Float16)(x3 * inv);
        *(v4h*)&qb[r * 128 + lane * 4] = qv;
    }
    __syncthreads();

    v8f acc[8] = {};
    #pragma unroll
    for (int kc = 0; kc < 4; ++kc) {
        const _Float16* arow = &qb[(m0 + laneM) * 128 + kc * 32];
        v8h alo = *(const v8h*)(arow + hiSel * 16);
        v8h ahi = *(const v8h*)(arow + hiSel * 16 + 8);
        v16h afrag;
        #pragma unroll
        for (int i = 0; i < 8; ++i) { afrag[i] = alo[i]; afrag[i + 8] = ahi[i]; }

        v16h bfr[8];
        #pragma unroll
        for (int nt = 0; nt < 8; ++nt) {
            const _Float16* brow = &cb[(nt * 16 + laneM) * 128 + kc * 32 + hiSel * 16];
            v8h b0 = *(const v8h*)(brow);
            v8h b1 = *(const v8h*)(brow + 8);
            #pragma unroll
            for (int i = 0; i < 8; ++i) { bfr[nt][i] = b0[i]; bfr[nt][i + 8] = b1[i]; }
        }
        #pragma unroll
        for (int nt = 0; nt < 8; ++nt) {
            acc[nt] = __builtin_amdgcn_wmma_f32_16x16x32_f16(
                false, afrag, false, bfr[nt], (short)0, acc[nt], false, false);
        }
    }

    float* obase = outg + (((size_t)b * EA_S + s0) * EA_H + h) * EA_D;
    #pragma unroll
    for (int nt = 0; nt < 8; ++nt) {
        #pragma unroll
        for (int r = 0; r < 8; ++r) {
            int row = m0 + r + hiSel * 8;
            obase[(size_t)row * (EA_H * EA_D) + nt * 16 + laneM] = acc[nt][r];
        }
    }
}

// ---------------------------------------------------------------------------
extern "C" void kernel_launch(void* const* d_in, const int* in_sizes, int n_in,
                              void* d_out, int out_size, void* d_ws, size_t ws_size,
                              hipStream_t stream) {
    const float* q = (const float*)d_in[0];
    const float* k = (const float*)d_in[1];
    const float* v = (const float*)d_in[2];
    float* out = (float*)d_out;
    float* ws  = (float*)d_ws;

    float* smax = ws;                       // B*H*D = 16384 floats
    float* sinv = ws + 16384;               // 16384 floats
    float* ctx  = ws + 32768;               // B*H*128*128 = 2,097,152 floats (8MB)

    ea_stats<<<EA_B * EA_H, 1024, 0, stream>>>(k, smax, sinv);
    ea_context<<<EA_B * EA_H, 256, 0, stream>>>(k, v, smax, sinv, ctx);
    dim3 g2(EA_S / 128, EA_B * EA_H);
    ea_out<<<g2, 256, 0, stream>>>(q, ctx, out);
}
